// cross_attention_24678882083069
// MI455X (gfx1250) — matrix operand order repaired and checked by reading, not compiled, not run on hardware
//
#include <hip/hip_runtime.h>
#include <hip/hip_bf16.h>

// Cross-attention for MI455X (gfx1250): bf16 WMMA everywhere, flash-attention
// second stage. Workspace: bf16 Q|K|V as [3][B][L][C] = 24 MB.
// All WMMA fragments are fed by contiguous 16-byte per-lane loads; the
// attention K-chunk is DMA'd into LDS by the Tensor Data Mover (TDM) with
// hardware row padding matching the 72-short LDS pitch.

#define Bn 4
#define Cd 512
#define Ld 2048
#define Hh 8
#define Dh 64

typedef __attribute__((ext_vector_type(16))) __bf16         v16bf;
typedef __attribute__((ext_vector_type(8)))  float          v8f;
typedef __attribute__((ext_vector_type(4)))  float          v4f;
typedef __attribute__((ext_vector_type(8)))  unsigned short v8us;
typedef __attribute__((ext_vector_type(4)))  unsigned int   v4u;
typedef __attribute__((ext_vector_type(8)))  int            v8i;
typedef __attribute__((ext_vector_type(4)))  int            v4i;

union Frag16 { v16bf v; v8us h[2]; };

// Hardware f32->bf16 convert (RNE); gfx1250 has native bf16 cvt.
__device__ inline unsigned short f2bf(float f) {
    return __builtin_bit_cast(unsigned short, (__bf16)f);
}
// Two converts fused into one v_cvt_pk_bf16_f32: lo -> [15:0], hi -> [31:16]
__device__ inline unsigned int pack2bf(float lo, float hi) {
    return (unsigned int)f2bf(lo) | ((unsigned int)f2bf(hi) << 16);
}
__device__ inline v16bf ld_frag(const unsigned short* p0,
                                const unsigned short* p1) {
    Frag16 f;
    f.h[0] = *(const v8us*)p0;                // 16B -> ds/global _load_b128
    f.h[1] = *(const v8us*)p1;
    return f.v;
}
__device__ inline v8f wmma_bf16(v16bf a, v16bf b, v8f c) {
    return __builtin_amdgcn_wmma_f32_16x16x32_bf16(
        /*neg_a=*/false, a, /*neg_b=*/false, b,
        /*c_mod=*/(short)0, c, /*reuse_a=*/false, /*reuse_b=*/false);
}

#if __has_builtin(__builtin_amdgcn_tensor_load_to_lds)
#define USE_TDM 1
#else
#define USE_TDM 0
#endif

#if USE_TDM
// TDM: DMA a 64x64 bf16 tile (row stride Cd elements) from global into LDS,
// inserting 16B of padding after each 128B row (-> 144B pitch = 72 shorts).
// D# per CDNA5 ISA §8.3/8.4; 2D tensor so groups 2/3 are zero.
// This toolchain's builtin takes 6 args (clang-23 / therock-10.0 form).
__device__ inline void tdm_load_k(const unsigned short* src,
                                  unsigned int lds_addr) {
    unsigned long long ga = (unsigned long long)(uintptr_t)src;
    v4u g0;
    g0[0] = 1u;                                  // count=1, user mode
    g0[1] = lds_addr;                            // LDS byte address
    g0[2] = (unsigned int)ga;                    // global_addr[31:0]
    g0[3] = (unsigned int)((ga >> 32) & 0x1FFFFFFu) | (2u << 30); // [56:32]|type=2
    v8i g1;
    g1[0] = (int)((1u << 16)       // data_size = 1 (2 bytes)
                | (1u << 20)       // pad_enable
                | (4u << 22)       // pad_interval: 32 DWORDs (one 64-elem row)
                | (3u << 25));     // pad_amount: 4 DWORDs (16 bytes)
    g1[1] = 0;                                   // no atomic barrier; dim0 lo=0
    g1[2] = (int)0x10u;                          // tensor_dim0 = 1<<20 (no OOB)
    g1[3] = (int)(0x10u | (64u << 16));          // tensor_dim1 hi; tile_dim0=64
    g1[4] = 64;                                  // tile_dim1=64, tile_dim2=0
    g1[5] = (int)Cd;                             // tensor_dim0_stride = 512
    g1[6] = 0;
    g1[7] = 0;
    v4i z4 = {0, 0, 0, 0};
    v8i z8 = {0, 0, 0, 0, 0, 0, 0, 0};
    __builtin_amdgcn_tensor_load_to_lds(g0, g1, z4, z4, z8, 0);
}
#endif

// ---------------------------------------------------------------------------
// Kernel 1: QKV projection.  Y[b, l, n] = sum_k x[b, k, l] * W[n, k] + bias[n]
// Output bf16 [L, C] row-major per (matrix, batch).
// Block tile 128(M) x 64(N): wave w owns M-subtile w and all 4 N-subtiles,
// so one A fragment feeds 4 WMMAs per K-step.
// grid = (L/128, C/64, 3*B), block = 256 (8 waves).
// ---------------------------------------------------------------------------
__global__ __launch_bounds__(256) void qkv_proj_kernel(
    const float* __restrict__ x1, const float* __restrict__ x2,
    const float* __restrict__ Wq, const float* __restrict__ bq,
    const float* __restrict__ Wk, const float* __restrict__ bk,
    const float* __restrict__ Wv, const float* __restrict__ bv,
    unsigned short* __restrict__ qkv)
{
    __shared__ alignas(16) unsigned short Xs[128][40];  // [m][k] bf16, padded
    __shared__ alignas(16) unsigned short Ws[64][40];   // [n][k] bf16, padded

    const int tid  = threadIdx.x;
    const int lane = tid & 31;
    const int w    = tid >> 5;
    const int l0   = blockIdx.x * 128;
    const int n0   = blockIdx.y * 64;
    const int z    = blockIdx.z;
    const int mat  = z % 3;                 // 0=Q, 1=K, 2=V
    const int bb   = z / 3;

    const float* xsrc = (mat == 0) ? x1 : x2;
    const float* W    = (mat == 0) ? Wq : (mat == 1) ? Wk : Wv;
    const float* bias = (mat == 0) ? bq : (mat == 1) ? bk : bv;
    const float* xb   = xsrc + (size_t)bb * Cd * Ld;
    unsigned short* dst = qkv + ((size_t)mat * Bn + bb) * Ld * Cd;

    const int col  = lane & 15;
    const int half = lane >> 4;

    float biasv[4];
    #pragma unroll
    for (int ns = 0; ns < 4; ++ns) biasv[ns] = bias[n0 + ns * 16 + col];

    v8f acc[4] = {v8f{}, v8f{}, v8f{}, v8f{}};

    // Staging assignment: two adjacent k-rows per thread so the transposed
    // LDS writes are packed b32 pairs.
    const int skp = (tid & 15) * 2;     // k pair base (0..30)
    const int sm0 = (tid >> 4) * 8;     // 8 m values   (0..120)

    for (int kk = 0; kk < Cd; kk += 32) {
        // Stage X transposed: Xs[m][k] = bf16(x[b, kk+k, l0+m]).
        {
            const float* r0 = xb + (size_t)(kk + skp) * Ld + l0 + sm0;
            const float* r1 = r0 + Ld;
            v4f a0 = *(const v4f*)r0,       a1 = *(const v4f*)(r0 + 4);
            v4f b0 = *(const v4f*)r1,       b1 = *(const v4f*)(r1 + 4);
            #pragma unroll
            for (int j = 0; j < 4; ++j) {
                *(unsigned int*)&Xs[sm0 + j][skp]     = pack2bf(a0[j], b0[j]);
                *(unsigned int*)&Xs[sm0 + 4 + j][skp] = pack2bf(a1[j], b1[j]);
            }
        }
        // Stage W: Ws[n][k] = bf16(W[(n0+n)*C + kk+k]); packed ds_store_b128.
        {
            int n = tid >> 2, k0 = (tid & 3) * 8;
            const float* src = W + (size_t)(n0 + n) * Cd + kk + k0;
            v4f f0 = *(const v4f*)src;
            v4f f1 = *(const v4f*)(src + 4);
            v8us pk;
            #pragma unroll
            for (int j = 0; j < 4; ++j) {
                pk[j]     = f2bf(f0[j]);
                pk[4 + j] = f2bf(f1[j]);
            }
            *(v8us*)&Ws[n][k0] = pk;
        }
        __syncthreads();

        // A fragment 16x32 (2x ds_load_b128) feeds 4 WMMAs
        const unsigned short* xr = Xs[w * 16 + col];
        v16bf a = ld_frag(xr + half * 8, xr + 16 + half * 8);

        #pragma unroll
        for (int ns = 0; ns < 4; ++ns) {
            const unsigned short* wr = Ws[ns * 16 + col];
            v16bf fb = ld_frag(wr + half * 8, wr + 16 + half * 8);
            acc[ns] = wmma_bf16(a, fb, acc[ns]);
        }
        __syncthreads();
    }

    // Epilogue: add bias, store bf16 [L, C]
    #pragma unroll
    for (int ns = 0; ns < 4; ++ns) {
        int n = n0 + ns * 16 + col;
        #pragma unroll
        for (int r = 0; r < 8; ++r) {
            int ml = w * 16 + r + 8 * half;
            dst[(size_t)(l0 + ml) * Cd + n] = f2bf(acc[ns][r] + biasv[ns]);
        }
    }
}

// ---------------------------------------------------------------------------
// Kernel 2: flash attention.  grid = (L/128, H, B), block = 256 (8 waves).
// Each wave owns 16 query rows; block shares 64-key K/V chunks in LDS.
// K chunk is TDM-DMA'd ([k][d], hardware-padded pitch); V staged transposed
// [d][k] with packed-pair b32 scatter.  Softmax in base-2 domain (v_exp_f32).
// ---------------------------------------------------------------------------
__global__ __launch_bounds__(256) void attn_kernel(
    const unsigned short* __restrict__ qkv, float* __restrict__ out)
{
    __shared__ alignas(16) unsigned short Ks[64][72];     // [k][d] bf16
    __shared__ alignas(16) unsigned short Vt[64][72];     // [d][k] bf16
    __shared__ alignas(16) unsigned short Ps[8][16][72];  // per-wave P slab

    const int tid  = threadIdx.x;
    const int lane = tid & 31;
    const int w    = tid >> 5;
    const int col  = lane & 15;
    const int half = lane >> 4;

    const int q0 = blockIdx.x * 128 + w * 16;
    const int h  = blockIdx.y;
    const int bb = blockIdx.z;

    const unsigned short* qb = qkv + ((size_t)0 * Bn + bb) * Ld * Cd;
    const unsigned short* kg = qkv + ((size_t)1 * Bn + bb) * Ld * Cd;
    const unsigned short* vg = qkv + ((size_t)2 * Bn + bb) * Ld * Cd;

#if USE_TDM
    const unsigned int ks_lds = (unsigned int)(uintptr_t)&Ks[0][0];
#endif

    // Q A-fragments: 4x global_load_b128, held across the whole K loop
    const unsigned short* qrow = qb + (size_t)(q0 + col) * Cd + h * Dh;
    v16bf qa0 = ld_frag(qrow + half * 8,      qrow + 16 + half * 8);
    v16bf qa1 = ld_frag(qrow + 32 + half * 8, qrow + 48 + half * 8);

    v8f acc[4] = {v8f{}, v8f{}, v8f{}, v8f{}};
    float mrow[8], rsum[8];
    #pragma unroll
    for (int r = 0; r < 8; ++r) { mrow[r] = -1e30f; rsum[r] = 0.0f; }

    // 1/sqrt(64) * log2(e): softmax computed with exp2
    const float SC = 0.125f * 1.44269504088896340736f;

    // Staging: two adjacent key rows per thread -> packed b32 V scatter.
    const int skr = (tid >> 3) * 2;    // key row pair base (0..62)
    const int sd0 = (tid & 7) * 8;     // 8 contiguous d values

    for (int k0i = 0; k0i < Ld; k0i += 64) {
        __syncthreads();
#if USE_TDM
        // K: one TDM descriptor DMAs the whole 64x64 tile into Ks (wave 0
        // issues; TDM ignores EXEC and issues once per wave).
        if (w == 0) tdm_load_k(kg + (size_t)k0i * Cd + h * Dh, ks_lds);
#endif
        {
            size_t off0 = (size_t)(k0i + skr) * Cd + h * Dh + sd0;
            size_t off1 = off0 + Cd;
#if !USE_TDM
            *(v8us*)&Ks[skr][sd0]     = *(const v8us*)(kg + off0);
            *(v8us*)&Ks[skr + 1][sd0] = *(const v8us*)(kg + off1);
#endif
            // V: b128 global loads -> packed-pair b32 scatter into Vt[d][k]
            v8us v0 = *(const v8us*)(vg + off0);
            v8us v1 = *(const v8us*)(vg + off1);
            #pragma unroll
            for (int j = 0; j < 8; ++j) {
                *(unsigned int*)&Vt[sd0 + j][skr] =
                    (unsigned int)v0[j] | ((unsigned int)v1[j] << 16);
            }
        }
        if (k0i + 64 < Ld) {   // global_prefetch_b8 of the next V chunk
            size_t poff = (size_t)(k0i + 64 + skr) * Cd + h * Dh + sd0;
            __builtin_prefetch(vg + poff, 0, 1);
        }
#if USE_TDM
        if (w == 0) __builtin_amdgcn_s_wait_tensorcnt(0);
#endif
        __syncthreads();

        // S[16q x 64k] = Q(16x64) @ K^T(64x64): 4 N-subtiles x 2 K-steps
        v8f s[4] = {v8f{}, v8f{}, v8f{}, v8f{}};
        #pragma unroll
        for (int dstep = 0; dstep < 2; ++dstep) {
            v16bf qa = dstep ? qa1 : qa0;
            #pragma unroll
            for (int ksub = 0; ksub < 4; ++ksub) {
                const unsigned short* krp = Ks[ksub * 16 + col];
                int d = dstep * 32 + half * 8;
                v16bf fb = ld_frag(krp + d, krp + d + 16);
                s[ksub] = wmma_bf16(qa, fb, s[ksub]);
            }
        }

        // Online softmax (base-2). Row m = r + 8*half lives in one 16-lane
        // half; shfl_xor masks 1/2/4/8 never cross the half boundary.
        #pragma unroll
        for (int r = 0; r < 8; ++r) {
            float a0 = s[0][r] * SC, a1 = s[1][r] * SC;
            float a2 = s[2][r] * SC, a3 = s[3][r] * SC;
            float mx = fmaxf(fmaxf(a0, a1), fmaxf(a2, a3));
            mx = fmaxf(mx, __shfl_xor(mx, 1, 32));
            mx = fmaxf(mx, __shfl_xor(mx, 2, 32));
            mx = fmaxf(mx, __shfl_xor(mx, 4, 32));
            mx = fmaxf(mx, __shfl_xor(mx, 8, 32));
            float mnew  = fmaxf(mrow[r], mx);
            float alpha = exp2f(mrow[r] - mnew);
            float p0 = exp2f(a0 - mnew);
            float p1 = exp2f(a1 - mnew);
            float p2 = exp2f(a2 - mnew);
            float p3 = exp2f(a3 - mnew);
            float ps = (p0 + p1) + (p2 + p3);
            ps += __shfl_xor(ps, 1, 32);
            ps += __shfl_xor(ps, 2, 32);
            ps += __shfl_xor(ps, 4, 32);
            ps += __shfl_xor(ps, 8, 32);
            rsum[r] = rsum[r] * alpha + ps;
            mrow[r] = mnew;
            #pragma unroll
            for (int d = 0; d < 4; ++d) acc[d][r] = acc[d][r] * alpha;
            int mq = r + 8 * half;
            Ps[w][mq][col]      = f2bf(p0);     // C-layout -> LDS
            Ps[w][mq][col + 16] = f2bf(p1);
            Ps[w][mq][col + 32] = f2bf(p2);
            Ps[w][mq][col + 48] = f2bf(p3);
        }

        // Re-read P in A-layout (same wave; LDS in-order, no barrier needed)
        const unsigned short* pr = Ps[w][col];
        v16bf pa0 = ld_frag(pr + half * 8,      pr + 16 + half * 8);
        v16bf pa1 = ld_frag(pr + 32 + half * 8, pr + 48 + half * 8);

        // O[16q x 64d] += P(16x64) @ V(64x64): 4 N-subtiles x 2 K-steps
        #pragma unroll
        for (int dsub = 0; dsub < 4; ++dsub) {
            const unsigned short* vr = Vt[dsub * 16 + col];
            v16bf fv0 = ld_frag(vr + half * 8,      vr + 16 + half * 8);
            acc[dsub] = wmma_bf16(pa0, fv0, acc[dsub]);
            v16bf fv1 = ld_frag(vr + 32 + half * 8, vr + 48 + half * 8);
            acc[dsub] = wmma_bf16(pa1, fv1, acc[dsub]);
        }
    }

    // Finalize. Rows r = 0..7 map to contiguous l, so pack 2x b128 stores.
    float rinv[8];
    #pragma unroll
    for (int r = 0; r < 8; ++r) rinv[r] = 1.0f / rsum[r];

    #pragma unroll
    for (int dsub = 0; dsub < 4; ++dsub) {
        int c = h * Dh + dsub * 16 + col;
        float* op = out + ((size_t)bb * Cd + c) * Ld + q0 + 8 * half;
        v4f o0, o1;
        #pragma unroll
        for (int r = 0; r < 4; ++r) {
            o0[r] = acc[dsub][r]     * rinv[r];
            o1[r] = acc[dsub][4 + r] * rinv[4 + r];
        }
        *(v4f*)op       = o0;
        *(v4f*)(op + 4) = o1;
    }
}

// ---------------------------------------------------------------------------
extern "C" void kernel_launch(void* const* d_in, const int* in_sizes, int n_in,
                              void* d_out, int out_size, void* d_ws, size_t ws_size,
                              hipStream_t stream) {
    const float* x1 = (const float*)d_in[0];
    const float* x2 = (const float*)d_in[1];
    const float* Wq = (const float*)d_in[2];
    const float* bq = (const float*)d_in[3];
    const float* Wk = (const float*)d_in[4];
    const float* bk = (const float*)d_in[5];
    const float* Wv = (const float*)d_in[6];
    const float* bv = (const float*)d_in[7];

    // bf16 Q|K|V workspace: 3 * B * L * C * 2 bytes = 24 MB
    unsigned short* qkv = (unsigned short*)d_ws;
    (void)ws_size; (void)in_sizes; (void)n_in; (void)out_size;

    dim3 pgrid(Ld / 128, Cd / 64, 3 * Bn);
    qkv_proj_kernel<<<pgrid, dim3(256), 0, stream>>>(x1, x2, Wq, bq, Wk, bk,
                                                     Wv, bv, qkv);

    dim3 agrid(Ld / 128, Hh, Bn);
    attn_kernel<<<agrid, dim3(256), 0, stream>>>(qkv, (float*)d_out);
}
